// FPModule_30631706755378
// MI455X (gfx1250) — matrix operand order repaired and checked by reading, not compiled, not run on hardware
//
#include <hip/hip_runtime.h>
#include <hip/hip_bf16.h>

typedef __attribute__((ext_vector_type(16))) _Float16 v16h;
typedef __attribute__((ext_vector_type(8)))  _Float16 v8h;
typedef __attribute__((ext_vector_type(8)))  float    v8f;

#define B_    16
#define NC    1024
#define NF    4096
#define CC    512
#define CS    256
#define HH    512
#define KCAT  768            // CC + CS
#define NFINE (B_ * NF)      // 65536

#define ASTRIDE_H 56         // LDS row stride in halves (112 B = 28 DWORDs, 16B-aligned, conflict-free)

// ---------------------------------------------------------------------------
// Repack a row-major fp32 weight matrix W[K][N] into f16 WMMA B-fragments.
//   k = kt*32 + (L>>4)*16 + h,  n = nt*16 + (L&15)
// stored contiguously: Wp[((kt*NT + nt)*32 + L)*16 + h]
// ---------------------------------------------------------------------------
__global__ void pack_w_kernel(const float* __restrict__ W, _Float16* __restrict__ Wp,
                              int K, int N) {
    int idx = blockIdx.x * blockDim.x + threadIdx.x;
    if (idx >= K * N) return;
    int h    = idx & 15;
    int L    = (idx >> 4) & 31;
    int tile = idx >> 9;
    int NT   = N >> 4;
    int nt   = tile % NT;
    int kt   = tile / NT;
    int k = kt * 32 + (L >> 4) * 16 + h;
    int n = nt * 16 + (L & 15);
    Wp[idx] = (_Float16)W[(size_t)k * N + n];
}

// ---------------------------------------------------------------------------
// KNN(k=3) interpolation + skip concat. One wave32 per fine point.
// ---------------------------------------------------------------------------
__global__ __launch_bounds__(256)
void knn_interp_kernel(const float* __restrict__ x, const float* __restrict__ pos,
                       const float* __restrict__ pos_skip, const float* __restrict__ x_skip,
                       _Float16* __restrict__ hcat) {
    int gtid = blockIdx.x * blockDim.x + threadIdx.x;
    int f    = gtid >> 5;            // fine point index (0..65535)
    int lane = threadIdx.x & 31;
    if (f >= NFINE) return;
    int b = f / NF;

    const float px = pos_skip[(size_t)f * 3 + 0];
    const float py = pos_skip[(size_t)f * 3 + 1];
    const float pz = pos_skip[(size_t)f * 3 + 2];

    float bd0 = 1e30f, bd1 = 1e30f, bd2 = 1e30f;
    int   bi0 = 0,     bi1 = 0,     bi2 = 0;
    const float* pcb = pos + (size_t)b * NC * 3;
    for (int c = lane; c < NC; c += 32) {
        float dx = pcb[c * 3 + 0] - px;
        float dy = pcb[c * 3 + 1] - py;
        float dz = pcb[c * 3 + 2] - pz;
        float d2 = dx * dx + dy * dy + dz * dz;
        if (d2 < bd2) {
            if (d2 < bd1) {
                bd2 = bd1; bi2 = bi1;
                if (d2 < bd0) { bd1 = bd0; bi1 = bi0; bd0 = d2; bi0 = c; }
                else          { bd1 = d2;  bi1 = c; }
            } else { bd2 = d2; bi2 = c; }
        }
    }

    float w0 = 0.f, w1 = 0.f, w2 = 0.f;
    int   g0 = 0,   g1 = 0,   g2 = 0;
    #pragma unroll
    for (int s = 0; s < 3; ++s) {
        float cd = bd0; int ci = bi0;
        #pragma unroll
        for (int off = 16; off > 0; off >>= 1) {
            float od = __shfl_xor(cd, off, 32);
            int   oi = __shfl_xor(ci, off, 32);
            if (od < cd || (od == cd && oi < ci)) { cd = od; ci = oi; }
        }
        float w = 1.0f / (sqrtf(cd) + 1e-8f);
        if (s == 0)      { w0 = w; g0 = ci; }
        else if (s == 1) { w1 = w; g1 = ci; }
        else             { w2 = w; g2 = ci; }
        if (bd0 == cd && bi0 == ci) {
            bd0 = bd1; bi0 = bi1;
            bd1 = bd2; bi1 = bi2;
            bd2 = 1e30f;
        }
    }

    float inv = 1.0f / (w0 + w1 + w2 + 1e-8f);
    const float* r0 = x + ((size_t)b * NC + g0) * CC;
    const float* r1 = x + ((size_t)b * NC + g1) * CC;
    const float* r2 = x + ((size_t)b * NC + g2) * CC;
    _Float16* orow = hcat + (size_t)f * KCAT;
    for (int ch = lane; ch < CC; ch += 32) {
        float v = (w0 * r0[ch] + w1 * r1[ch] + w2 * r2[ch]) * inv;
        orow[ch] = (_Float16)v;
    }
    const float* srow = x_skip + (size_t)f * CS;
    for (int ch = lane; ch < CS; ch += 32) {
        orow[CC + ch] = (_Float16)srow[ch];
    }
}

// ---------------------------------------------------------------------------
// Issue async global->LDS copy of one 128x32(f16) A tile for k-step kt.
// 256 threads x 2 x b128 (16B) = 8KB. LDS row stride = 112B (padded).
// ---------------------------------------------------------------------------
__device__ __forceinline__ void async_load_a_tile(const _Float16* A, int blockM, int K,
                                                  int kt, unsigned ldsbase, int tid) {
    int s  = tid & 3;            // 16B sub-chunk within the 64B row slice
    int r0 = tid >> 2;           // rows 0..63
    int r1 = r0 + 64;            // rows 64..127
    unsigned l0 = ldsbase + (unsigned)(r0 * (ASTRIDE_H * 2) + s * 16);
    unsigned l1 = ldsbase + (unsigned)(r1 * (ASTRIDE_H * 2) + s * 16);
    unsigned long long g0 = (unsigned long long)(uintptr_t)
        (A + (size_t)(blockM + r0) * K + kt * 32 + s * 8);
    unsigned long long g1 = (unsigned long long)(uintptr_t)
        (A + (size_t)(blockM + r1) * K + kt * 32 + s * 8);
    asm volatile("global_load_async_to_lds_b128 %0, %1, off"
                 :: "v"(l0), "v"(g0) : "memory");
    asm volatile("global_load_async_to_lds_b128 %0, %1, off"
                 :: "v"(l1), "v"(g1) : "memory");
}

// ---------------------------------------------------------------------------
// GEMM (M=65536, N=512, K param) with v_wmma_f32_16x16x32_f16, bias + ReLU.
// Block = 256 threads = 8 waves (2 x 4).  Block tile 128x128, wave tile 64x32.
// A staged in LDS via async copies (double buffered); B pre-packed in global.
// ---------------------------------------------------------------------------
template<bool OUT_F16>
__global__ __launch_bounds__(256)
void gemm_bias_relu_kernel(const _Float16* __restrict__ A, const _Float16* __restrict__ Wp,
                           const float* __restrict__ bias, void* __restrict__ out, int K) {
    __shared__ _Float16 abuf[2][128 * ASTRIDE_H];   // 2 x 14KB

    const int N  = 512;
    const int NT = N / 16;               // 32 n-tiles
    int tid  = threadIdx.x;
    int lane = tid & 31;
    int w    = tid >> 5;                 // wave 0..7
    int wm   = w & 1;                    // 2 waves along M
    int wn   = w >> 1;                   // 4 waves along N
    int blockM = blockIdx.x * 128;
    int blockN = blockIdx.y * 128;
    int laneHalf = lane >> 4;            // 0 or 1
    int lmod     = lane & 15;

    unsigned sbase = (unsigned)(size_t)(&abuf[0][0]);   // LDS byte address
    const unsigned bufBytes = 128 * ASTRIDE_H * 2;

    v8f zero;
    #pragma unroll
    for (int r = 0; r < 8; ++r) zero[r] = 0.0f;

    v8f acc[4][2];
    #pragma unroll
    for (int i = 0; i < 4; ++i)
        #pragma unroll
        for (int j = 0; j < 2; ++j)
            acc[i][j] = zero;

    const int nsteps = K >> 5;           // K / 32
    const int ntBase = (blockN >> 4) + wn * 2;

    // preload k-step 0 into buffer 0
    async_load_a_tile(A, blockM, K, 0, sbase, tid);
    asm volatile("s_wait_asynccnt 0x0" ::: "memory");
    __syncthreads();

    for (int kt = 0; kt < nsteps; ++kt) {
        int cur = kt & 1;
        if (kt + 1 < nsteps)
            async_load_a_tile(A, blockM, K, kt + 1, sbase + (cur ^ 1) * bufBytes, tid);

        // B fragments: one contiguous 32B load per lane per tile (L2-resident)
        v16h bfrag[2];
        #pragma unroll
        for (int j = 0; j < 2; ++j) {
            size_t fo = (((size_t)kt * NT + (ntBase + j)) * 32 + lane) * 16;
            bfrag[j] = *(const v16h*)(Wp + fo);
        }
        // A fragments from LDS: two ds_load_b128 per lane per 16-row subtile
        const _Float16* ab = &abuf[cur][0];
        v16h afrag[4];
        #pragma unroll
        for (int i = 0; i < 4; ++i) {
            int m = wm * 64 + i * 16 + lmod;
            union { v16h v; v8h h[2]; } u;
            u.h[0] = *(const v8h*)(ab + m * ASTRIDE_H + laneHalf * 16);
            u.h[1] = *(const v8h*)(ab + m * ASTRIDE_H + laneHalf * 16 + 8);
            afrag[i] = u.v;
        }
        #pragma unroll
        for (int i = 0; i < 4; ++i)
            #pragma unroll
            for (int j = 0; j < 2; ++j)
                acc[i][j] = __builtin_amdgcn_wmma_f32_16x16x32_f16(
                    /*neg_a=*/false, afrag[i], /*neg_b=*/false, bfrag[j],
                    /*c_mod=*/(short)0, acc[i][j],
                    /*reuse_a=*/false, /*reuse_b=*/false);

        asm volatile("s_wait_asynccnt 0x0" ::: "memory");
        __syncthreads();
    }

    // epilogue: bias + ReLU.  C/D layout: VGPR r -> M = r + (lane/16)*8, N = lane%16
    #pragma unroll
    for (int i = 0; i < 4; ++i) {
        #pragma unroll
        for (int j = 0; j < 2; ++j) {
            int mbase = blockM + wm * 64 + i * 16 + laneHalf * 8;
            int nn    = blockN + wn * 32 + j * 16 + lmod;
            float bv = bias[nn];
            #pragma unroll
            for (int r = 0; r < 8; ++r) {
                float v = acc[i][j][r] + bv;
                v = v > 0.0f ? v : 0.0f;
                size_t o = (size_t)(mbase + r) * 512 + nn;
                if (OUT_F16) ((_Float16*)out)[o] = (_Float16)v;
                else         ((float*)out)[o] = v;
            }
        }
    }
}

// ---------------------------------------------------------------------------
// Launch: pack W1, pack W2, knn+concat, GEMM1 (relu, f16), GEMM2 (relu, f32)
// ---------------------------------------------------------------------------
extern "C" void kernel_launch(void* const* d_in, const int* in_sizes, int n_in,
                              void* d_out, int out_size, void* d_ws, size_t ws_size,
                              hipStream_t stream) {
    const float* x        = (const float*)d_in[0];
    const float* pos      = (const float*)d_in[1];
    // d_in[2] = batch (unused; uniform layout)
    const float* x_skip   = (const float*)d_in[3];
    const float* pos_skip = (const float*)d_in[4];
    // d_in[5] = batch_skip (unused)
    const float* W1 = (const float*)d_in[6];
    const float* b1 = (const float*)d_in[7];
    const float* W2 = (const float*)d_in[8];
    const float* b2 = (const float*)d_in[9];

    char* ws = (char*)d_ws;
    _Float16* hcat = (_Float16*)ws;                                     // 65536 x 768 f16
    _Float16* h1   = (_Float16*)(ws + (size_t)NFINE * KCAT * 2);        // 65536 x 512 f16
    _Float16* W1p  = (_Float16*)(ws + (size_t)NFINE * KCAT * 2
                                    + (size_t)NFINE * HH * 2);          // 768 x 512 f16
    _Float16* W2p  = W1p + (size_t)KCAT * HH;                           // 512 x 512 f16

    pack_w_kernel<<<(KCAT * HH + 255) / 256, 256, 0, stream>>>(W1, W1p, KCAT, HH);
    pack_w_kernel<<<(HH * HH + 255) / 256, 256, 0, stream>>>(W2, W2p, HH, HH);

    knn_interp_kernel<<<NFINE / 8, 256, 0, stream>>>(x, pos, pos_skip, x_skip, hcat);

    dim3 g(NFINE / 128, 4);
    gemm_bias_relu_kernel<true ><<<g, 256, 0, stream>>>(hcat, W1p, b1, (void*)h1, KCAT);
    gemm_bias_relu_kernel<false><<<g, 256, 0, stream>>>(h1,   W2p, b2, d_out,     HH);
}
